// LSTM_9113920602502
// MI455X (gfx1250) — matrix operand order repaired and checked by reading, not compiled, not run on hardware
//
#include <hip/hip_runtime.h>
#include <stdint.h>

// ---------------------------------------------------------------------------
// LSTM (B=4096, T=512, D=8, H=48, O=1) fused persistent kernel for gfx1250.
//   gates = [h | x_t | 1 | 0](16x64 bf16)  @  W_cat(64x192 bf16)  -> f32
//   via 2x v_wmma_f32_16x16x32_bf16 per 16x16 gate tile.
// Workgroup = 96 threads (3 waves) = one batch tile of 16 rows.
// Wave j owns hidden cols [16j,16j+16): gate tiles {j, 3+j, 6+j, 9+j}.
// Nonlinearities use native v_tanh_f32 when available (CDNA5 TRANS op).
// ---------------------------------------------------------------------------

typedef unsigned int u32;
typedef __attribute__((ext_vector_type(8)))  u32    u32x8;
typedef __attribute__((ext_vector_type(16))) __bf16 v16bf;
typedef __attribute__((ext_vector_type(8)))  float  v8f;

static constexpr int Bn = 4096, Tn = 512, Dn = 8, Hn = 48, Gn = 192; // 4H
static constexpr int NTILES = Gn / 16;        // 12
static constexpr int NFRAG  = NTILES * 2;     // 24 B-fragments (ntile, khalf)

static __device__ __forceinline__ v16bf mkv(uint4 a, uint4 b) {
    u32x8 t = {a.x, a.y, a.z, a.w, b.x, b.y, b.z, b.w};
    return __builtin_bit_cast(v16bf, t);
}

static __device__ __forceinline__ v8f wmma_bf16(v16bf a, v16bf b, v8f c) {
    return __builtin_amdgcn_wmma_f32_16x16x32_bf16(
        /*neg_a=*/false, a, /*neg_b=*/false, b,
        /*c_mod=*/(short)0, c, /*reuse_a=*/false, /*reuse_b=*/false);
}

static __device__ __forceinline__ uint16_t f2bf(float f) {
    u32 u = __builtin_bit_cast(u32, f);
    u32 r = u + 0x7FFFu + ((u >> 16) & 1u);   // round-to-nearest-even
    return (uint16_t)(r >> 16);
}
static __device__ __forceinline__ u32 pack2bf(float lo, float hi) {
    return (u32)f2bf(lo) | ((u32)f2bf(hi) << 16);
}

static __device__ __forceinline__ float fexp2(float x) {
#if __has_builtin(__builtin_amdgcn_exp2f)
    return __builtin_amdgcn_exp2f(x);
#else
    return exp2f(x);
#endif
}
static __device__ __forceinline__ float frcp(float x) {
#if __has_builtin(__builtin_amdgcn_rcpf)
    return __builtin_amdgcn_rcpf(x);
#else
    return 1.0f / x;
#endif
}

// tanh: native v_tanh_f32 on gfx1250 if clang exposes it; else (1-e)/(1+e).
static __device__ __forceinline__ float tanh_(float x) {
#if __has_builtin(__builtin_amdgcn_tanhf)
    return __builtin_amdgcn_tanhf(x);
#else
    float e = fexp2(-2.88539008177792681472f * x);
    return (1.0f - e) * frcp(1.0f + e);
#endif
}
// sigmoid: 0.5*tanh(0.5x)+0.5 (3 ops w/ native tanh), else exp2/rcp form.
static __device__ __forceinline__ float sigm(float x) {
#if __has_builtin(__builtin_amdgcn_tanhf)
    return __builtin_fmaf(0.5f, __builtin_amdgcn_tanhf(0.5f * x), 0.5f);
#else
    return frcp(1.0f + fexp2(-1.44269504088896340736f * x));
#endif
}

__global__ __launch_bounds__(96)
void lstm_fused_kernel(const float* __restrict__ x,
                       const float* __restrict__ W_ih,
                       const float* __restrict__ W_hh,
                       const float* __restrict__ b_ih,
                       const float* __restrict__ b_hh,
                       const float* __restrict__ W_fc,
                       const float* __restrict__ b_fc,
                       float* __restrict__ out) {
    // W_cat in pre-swizzled per-lane B-fragment layout (ISA 7.12.2):
    //   fragment f = ntile*2 + khalf ; lane chunk = 32B (16 bf16, 2 per dword)
    //   dword r of lane l holds K = khalf*32 + (l>=16?8:0) + 2r + (r>=4?8:0) (+1 in hi half),
    //   N = ntile*16 + (l&15).
    __shared__ uint4 sW4[NFRAG * 32 * 2];   // 24576 B
    // A staging: 16 rows x 64 cols bf16, row-major (128 B per row).
    __shared__ uint4 sA4[16 * 8];           // 2048 B
    __shared__ float sHf[16 * Hn];          // final h, f32

    u32*      sWdw = reinterpret_cast<u32*>(sW4);
    u32*      sAdw = reinterpret_cast<u32*>(sA4);
    uint16_t* sA16 = reinterpret_cast<uint16_t*>(sA4);

    const int tid   = threadIdx.x;        // 0..95
    const int lane  = tid & 31;
    const int laneN = lane & 15;
    const int hi    = (lane >> 4) & 1;
    const int j     = tid >> 5;           // wave id 0..2 = hidden-col group
    const int tile  = blockIdx.x;         // batch tile, 16 rows
    const int b0    = tile * 16;

    // ---- one-time init: swizzle W_cat into LDS --------------------------
    for (int idx = tid; idx < NFRAG * 32 * 8; idx += 96) {
        int frag = idx >> 8;              // /256 dwords per fragment
        int rem  = idx & 255;
        int l    = rem >> 3;              // lane slot 0..31
        int r    = rem & 7;               // dword in 32B chunk
        int ntile = frag >> 1, khalf = frag & 1;
        int N = ntile * 16 + (l & 15);
        int K = khalf * 32 + ((l >= 16) ? 8 : 0) + 2 * r + ((r >= 4) ? 8 : 0);
        float v[2];
        #pragma unroll
        for (int q = 0; q < 2; ++q) {
            int k = K + q;
            float w;
            if      (k < 48) w = W_hh[N * Hn + k];
            else if (k < 56) w = W_ih[N * Dn + (k - 48)];
            else if (k == 56) w = b_ih[N] + b_hh[N];
            else w = 0.0f;
            v[q] = w;
        }
        sWdw[idx] = pack2bf(v[0], v[1]);
    }
    // zero A staging (h0 = 0, pad cols = 0)
    for (int idx = tid; idx < 16 * 32; idx += 96) sAdw[idx] = 0u;
    __syncthreads();
    if (tid < 16) {
        sA16[tid * 64 + 56] = (uint16_t)0x3F80;   // bias column = 1.0bf
        // stage x for t = 0
        const float4* xp = reinterpret_cast<const float4*>(
            x + ((size_t)(b0 + tid) * Tn + 0) * Dn);
        float4 xa = xp[0], xb = xp[1];
        uint4 p;
        p.x = pack2bf(xa.x, xa.y); p.y = pack2bf(xa.z, xa.w);
        p.z = pack2bf(xb.x, xb.y); p.w = pack2bf(xb.z, xb.w);
        sA4[tid * 8 + 6] = p;                     // cols 48..55
    }
    __syncthreads();

    float cst[8], hst[8];
    #pragma unroll
    for (int r = 0; r < 8; ++r) { cst[r] = 0.0f; hst[r] = 0.0f; }

    const int colh  = j * 16 + laneN;             // this wave's hidden cols
    const bool xdut = (tid < 16);

    for (int t = 0; t < Tn; ++t) {
        // ---- load A fragments (h_{t}, x_t, bias) from LDS ---------------
        const uint4* pa = &sA4[(laneN << 3) + hi];
        v16bf a0 = mkv(pa[0], pa[2]);             // K =  0..31
        v16bf a1 = mkv(pa[4], pa[6]);             // K = 32..63

        // ---- prefetch x_{t+1} (wave0 rows) while WMMAs run --------------
        float4 xa, xb;
        if (xdut) {
            int tt = (t + 1 < Tn) ? (t + 1) : (Tn - 1);
            const float4* xp = reinterpret_cast<const float4*>(
                x + ((size_t)(b0 + laneN) * Tn + tt) * Dn);
            xa = xp[0]; xb = xp[1];
        }

        // ---- 4 gate tiles: i, f, g, o (2 WMMAs each) --------------------
        auto gate_tile = [&](int ntile) -> v8f {
            const uint4* p0 = &sW4[(((ntile * 2 + 0) * 32 + lane) << 1)];
            const uint4* p1 = &sW4[(((ntile * 2 + 1) * 32 + lane) << 1)];
            v16bf bb0 = mkv(p0[0], p0[1]);
            v16bf bb1 = mkv(p1[0], p1[1]);
            v8f acc = {0.f, 0.f, 0.f, 0.f, 0.f, 0.f, 0.f, 0.f};
            acc = wmma_bf16(a0, bb0, acc);
            acc = wmma_bf16(a1, bb1, acc);
            return acc;
        };
        v8f gi = gate_tile(j);
        v8f gf = gate_tile(3 + j);
        v8f gg = gate_tile(6 + j);
        v8f go = gate_tile(9 + j);

        // ---- elementwise cell update (per-lane, D-layout) ---------------
        #pragma unroll
        for (int r = 0; r < 8; ++r) {
            float iv = sigm(gi[r]);
            float fv = sigm(gf[r]);
            float gv = tanh_(gg[r]);
            float ov = sigm(go[r]);
            float cv = fv * cst[r] + iv * gv;
            cst[r] = cv;
            hst[r] = ov * tanh_(cv);
        }

        // ---- feed h_{t+1} back into A staging (bf16) --------------------
        #pragma unroll
        for (int r = 0; r < 8; ++r)
            sA16[(r + 8 * hi) * 64 + colh] = f2bf(hst[r]);

        // ---- store staged x_{t+1} ---------------------------------------
        if (xdut) {
            uint4 p;
            p.x = pack2bf(xa.x, xa.y); p.y = pack2bf(xa.z, xa.w);
            p.z = pack2bf(xb.x, xb.y); p.w = pack2bf(xb.z, xb.w);
            sA4[laneN * 8 + 6] = p;
        }
        __syncthreads();
    }

    // ---- final fc: out[b] = h_last . W_fc + b_fc (O = 1) ----------------
    #pragma unroll
    for (int r = 0; r < 8; ++r)
        sHf[(r + 8 * hi) * Hn + colh] = hst[r];
    __syncthreads();
    if (tid < 16) {
        float s = b_fc[0];
        #pragma unroll 8
        for (int k = 0; k < Hn; ++k) s += sHf[tid * Hn + k] * W_fc[k];
        out[b0 + tid] = s;
    }
}

extern "C" void kernel_launch(void* const* d_in, const int* in_sizes, int n_in,
                              void* d_out, int out_size, void* d_ws, size_t ws_size,
                              hipStream_t stream) {
    const float* x    = (const float*)d_in[0];
    const float* W_ih = (const float*)d_in[1];
    const float* W_hh = (const float*)d_in[2];
    const float* b_ih = (const float*)d_in[3];
    const float* b_hh = (const float*)d_in[4];
    const float* W_fc = (const float*)d_in[5];
    const float* b_fc = (const float*)d_in[6];
    float* out = (float*)d_out;

    dim3 grid(Bn / 16);   // 256 batch tiles
    dim3 block(96);       // 3 waves: one per hidden-column group
    lstm_fused_kernel<<<grid, block, 0, stream>>>(x, W_ih, W_hh, b_ih, b_hh,
                                                  W_fc, b_fc, out);
}
